// BaselineModel_80796924772520
// MI455X (gfx1250) — matrix operand order repaired and checked by reading, not compiled, not run on hardware
//
#include <hip/hip_runtime.h>
#include <hip/hip_bf16.h>

typedef __attribute__((ext_vector_type(16))) __bf16 v16bf;
typedef __attribute__((ext_vector_type(8)))  __bf16 v8bf;
typedef __attribute__((ext_vector_type(8)))  float  v8f;

#define BM 128
#define BN 64
#define KT 64
#define PAr 80   // A LDS row pitch (elems): multiple of 16 -> 32B-aligned fragments
#define PBT 80   // B LDS column pitch (elems)

// ---------------- small prep kernels ----------------

// Fold conv bias + eval-mode BN into scale/shift: y = conv*s + shift
__global__ void bn_prep(const float* __restrict__ g, const float* __restrict__ b,
                        const float* __restrict__ m, const float* __restrict__ v,
                        const float* __restrict__ convb,
                        float* __restrict__ scale, float* __restrict__ shift, int n) {
  int i = blockIdx.x * blockDim.x + threadIdx.x;
  if (i < n) {
    float s = g[i] * rsqrtf(v[i] + 1e-5f);
    scale[i] = s;
    shift[i] = (convb[i] - m[i]) * s + b[i];
  }
}

// conv weight [O,C,3,3] f32 -> B matrix [N=O][K=9*C] bf16, K = kidx*C + c
__global__ void conv_w_prep(const float* __restrict__ w, __bf16* __restrict__ out,
                            int O, int C) {
  int i = blockIdx.x * blockDim.x + threadIdx.x;
  int total = O * C * 9;
  if (i < total) {
    int kidx = i % 9;
    int c    = (i / 9) % C;
    int o    = i / (9 * C);
    out[(size_t)o * (9 * C) + kidx * C + c] = (__bf16)w[i];
  }
}

// fc weight [N,K] f32 is already B-layout [N][K]: pure cast to bf16
__global__ void cast_bf16(const float* __restrict__ w, __bf16* __restrict__ out, int n) {
  int i = blockIdx.x * blockDim.x + threadIdx.x;
  if (i < n) out[i] = (__bf16)w[i];
}

// ---------------- conv1 (1 input channel): direct, fused BN+ReLU+pool ----------------
// grid.x = B*24*24 (one pooled pixel), 256 threads = 256 output channels.
// Output NHWC bf16 [B,24,24,256] for conv2's implicit GEMM.
__global__ __launch_bounds__(256)
void conv1_block_kernel(const float* __restrict__ x, const float* __restrict__ w,
                        const float* __restrict__ scale, const float* __restrict__ shift,
                        __bf16* __restrict__ act1) {
  __shared__ float xt[16];
  int pix = blockIdx.x;
  int pw = pix % 24;
  int ph = (pix / 24) % 24;
  int b  = pix / 576;
  int t  = threadIdx.x;
  if (t < 16) {
    int r = t >> 2, cc = t & 3;
    int ih = 2 * ph - 1 + r, iw = 2 * pw - 1 + cc;
    float v = 0.f;
    if (ih >= 0 && ih < 48 && iw >= 0 && iw < 48) v = x[b * 2304 + ih * 48 + iw];
    xt[t] = v;
  }
  __syncthreads();
  float wr[9];
#pragma unroll
  for (int k = 0; k < 9; ++k) wr[k] = w[t * 9 + k];
  float s = scale[t], sh = shift[t];
  float best = 0.f;  // ReLU outputs >= 0, so 0 is a valid identity for the max
#pragma unroll
  for (int dy = 0; dy < 2; ++dy)
#pragma unroll
    for (int dx = 0; dx < 2; ++dx) {
      float a = 0.f;
#pragma unroll
      for (int ky = 0; ky < 3; ++ky)
#pragma unroll
        for (int kx = 0; kx < 3; ++kx)
          a += xt[(dy + ky) * 4 + (dx + kx)] * wr[ky * 3 + kx];
      best = fmaxf(best, fmaxf(a * s + sh, 0.f));
    }
  act1[((size_t)(b * 24 + ph) * 24 + pw) * 256 + t] = (__bf16)best;
}

// A-matrix 32-K-block swizzle (involution on 8-elem groups): [0-7][16-23][8-15][24-31]
__device__ __forceinline__ int swz_group(int g) {  // g in 0..3
  return g ^ ((((g >> 1) ^ g) & 1) * 3);
}

// ---------------- WMMA GEMM: C = A @ B^T(B stored [N][K]), bf16 in, f32 acc ----------
// AMODE: 0 = dense row-major A [M,K]; 1 = implicit im2col from NHWC activation
//        (M enumerates conv-output pixels in pool-grouped order: b, window, 2x2-sub)
// EMODE: 0 = +bias, write bf16 (outb) and/or f32 (outf), row-major [M,N]
//        1 = BN+ReLU+2x2 maxpool, write NHWC bf16 [B,PH,PW,N]
//        2 = BN+ReLU+2x2 maxpool, write NCHW f32 AND bf16 [B,N,PH,PW]
template <int AMODE, int EMODE>
__global__ __launch_bounds__(256)
void gemm_wmma(const __bf16* __restrict__ A, const __bf16* __restrict__ Bm,
               const float* __restrict__ scale, const float* __restrict__ shift,
               __bf16* __restrict__ outb, float* __restrict__ outf,
               int M, int N, int K, int H, int W, int C, int PHp, int PWd) {
  __shared__ __align__(32) __bf16 As[BM * PAr];   // [row][plain K]   20.0 KB
  __shared__ __align__(32) __bf16 BsT[BN * PBT];  // [col][plain K]      10.0 KB
  const int t    = threadIdx.x;
  const int half = (t >> 4) & 1;  // lane half
  const int lr   = t & 15;
  const int wave = t >> 5;
  const int wm   = (wave >> 1) * 32;  // 4 waves down
  const int wn   = (wave & 1) * 32;   // 2 waves across
  const int m0blk = blockIdx.x * BM;
  const int n0blk = blockIdx.y * BN;

  // Pre-decode the 4 fixed A-tile rows this thread stages (once, outside K loop)
  int abase[4], hh[4], ww[4];
  if (AMODE == 1) {
    int hw = H * W;
#pragma unroll
    for (int i = 0; i < 4; ++i) {
      int idx = t + 256 * i;
      int r   = idx >> 3;
      int m   = m0blk + r;
      int b   = m / hw;
      int rem = m - b * hw;
      int win = rem >> 2, sub = rem & 3;
      int h = ((win / PWd) << 1) + (sub >> 1);
      int w = ((win % PWd) << 1) + (sub & 1);
      hh[i] = h;
      ww[i] = w;
      abase[i] = ((b * H + h) * W + w) * C;
    }
  }

  v8f acc[2][2];
  v8f zero = {0.f, 0.f, 0.f, 0.f, 0.f, 0.f, 0.f, 0.f};
  acc[0][0] = zero; acc[0][1] = zero; acc[1][0] = zero; acc[1][1] = zero;

  for (int k0 = 0; k0 < K; k0 += KT) {
    // ---- stage A tile [BM x KT], 16B chunks, plain K order ----
    if (AMODE == 0) {
#pragma unroll
      for (int i = 0; i < 4; ++i) {  // 128*64 elems / 8 per chunk / 256 threads
        int idx = t + 256 * i;
        int r   = idx >> 3;
        int kin = (idx & 7) << 3;
        v8bf val = *(const v8bf*)(A + (size_t)(m0blk + r) * K + k0 + kin);
        int dk = kin;
        *(v8bf*)&As[r * PAr + dk] = val;
      }
    } else {
      int kidx = k0 / C;  // KT divides C -> one (ky,kx) per K-tile
      int c0   = k0 - kidx * C;
      int ky   = kidx / 3 - 1, kx = kidx % 3 - 1;
      int koff = (ky * W + kx) * C + c0;
#pragma unroll
      for (int i = 0; i < 4; ++i) {
        int idx = t + 256 * i;
        int r   = idx >> 3;
        int kin = (idx & 7) << 3;
        int ih = hh[i] + ky, iw = ww[i] + kx;
        v8bf val = {};
        if (ih >= 0 && ih < H && iw >= 0 && iw < W)
          val = *(const v8bf*)(A + abase[i] + koff + kin);
        int dk = kin;
        *(v8bf*)&As[r * PAr + dk] = val;
      }
    }
    // ---- stage B tile [BN cols x KT], plain K order ----
#pragma unroll
    for (int i = 0; i < 2; ++i) {  // 64*64 elems / 8 / 256 threads
      int idx = t + 256 * i;
      int col = idx >> 3;
      int kin = (idx & 7) << 3;
      v8bf val = *(const v8bf*)(Bm + (size_t)(n0blk + col) * K + k0 + kin);
      *(v8bf*)&BsT[col * PBT + kin] = val;
    }
    __syncthreads();

#pragma unroll
    for (int ks = 0; ks < KT; ks += 32) {
      // A fragment: contiguous 32B, lane-half h holds K = h*16 + 0..15 in order, same as B
      v16bf a0 = *(const v16bf*)&As[(wm + lr) * PAr + ks + half * 16];
      v16bf a1 = *(const v16bf*)&As[(wm + 16 + lr) * PAr + ks + half * 16];
      // B fragment: lane = N column, lane-half h holds K = h*16 + 0..15 in order
      v16bf b0 = *(const v16bf*)&BsT[(wn + lr) * PBT + ks + half * 16];
      v16bf b1 = *(const v16bf*)&BsT[(wn + 16 + lr) * PBT + ks + half * 16];
      acc[0][0] = __builtin_amdgcn_wmma_f32_16x16x32_bf16(false, a0, false, b0,
                                                          (short)0, acc[0][0], false, false);
      acc[0][1] = __builtin_amdgcn_wmma_f32_16x16x32_bf16(false, a0, false, b1,
                                                          (short)0, acc[0][1], false, false);
      acc[1][0] = __builtin_amdgcn_wmma_f32_16x16x32_bf16(false, a1, false, b0,
                                                          (short)0, acc[1][0], false, false);
      acc[1][1] = __builtin_amdgcn_wmma_f32_16x16x32_bf16(false, a1, false, b1,
                                                          (short)0, acc[1][1], false, false);
    }
    __syncthreads();
  }

  // ---- epilogue ----
  const int Npool = PHp * PWd;
#pragma unroll
  for (int ti = 0; ti < 2; ++ti) {
#pragma unroll
    for (int tj = 0; tj < 2; ++tj) {
      int n = n0blk + wn + tj * 16 + lr;
      if (EMODE == 0) {
        float bias = shift[n];
#pragma unroll
        for (int j = 0; j < 8; ++j) {
          int m = m0blk + wm + ti * 16 + half * 8 + j;
          float vl = acc[ti][tj][j] + bias;
          if (outb) outb[(size_t)m * N + n] = (__bf16)vl;
          if (outf) outf[(size_t)m * N + n] = vl;
        }
      } else {
        float s = scale[n], sh = shift[n];
        // C layout: VGPR j holds row (half*8 + j); rows 4p..4p+3 = pooling window p.
        float p0 = 0.f, p1 = 0.f;
#pragma unroll
        for (int j = 0; j < 4; ++j) p0 = fmaxf(p0, fmaxf(acc[ti][tj][j] * s + sh, 0.f));
#pragma unroll
        for (int j = 4; j < 8; ++j) p1 = fmaxf(p1, fmaxf(acc[ti][tj][j] * s + sh, 0.f));
        int base = m0blk + wm + ti * 16;    // 16-aligned
        int g0   = (base >> 2) + half * 2;  // pooled-pixel index
#pragma unroll
        for (int pi = 0; pi < 2; ++pi) {
          float vv = pi ? p1 : p0;
          int g   = g0 + pi;
          int b   = g / Npool;
          int rem = g - b * Npool;
          int ph  = rem / PWd, pw = rem - ph * PWd;
          if (EMODE == 1) {
            outb[((size_t)(b * PHp + ph) * PWd + pw) * N + n] = (__bf16)vv;
          } else {
            size_t o = ((size_t)(b * N + n) * PHp + ph) * PWd + pw;
            outf[o] = vv;
            outb[o] = (__bf16)vv;
          }
        }
      }
    }
  }
}

// ---------------- attention + fc3 (per-batch, fp32) ----------------
// out3 f32 NCHW flat [B,64,36]; q f32 [B,64]; out f32 [B,7]
__global__ __launch_bounds__(64)
void attn_kernel(const float* __restrict__ out3, const float* __restrict__ q,
                 const float* __restrict__ fc3w, const float* __restrict__ fc3b,
                 float* __restrict__ out) {
  __shared__ float qs[64], sc[36], at[36], gs[64];
  int b = blockIdx.x;
  int t = threadIdx.x;
  qs[t] = q[b * 64 + t];
  __syncthreads();
  if (t < 36) {
    float s = 0.f;
    for (int c = 0; c < 64; ++c) s += out3[(size_t)(b * 64 + c) * 36 + t] * qs[c];
    sc[t] = s;
  }
  __syncthreads();
  float mx = -1e30f;
  for (int i = 0; i < 36; ++i) mx = fmaxf(mx, sc[i]);
  float sum = 0.f;
  for (int i = 0; i < 36; ++i) sum += __expf(sc[i] - mx);
  if (t < 36) at[t] = __expf(sc[t] - mx) / sum;
  __syncthreads();
  {
    float g = 0.f;
    for (int i = 0; i < 36; ++i) g += out3[(size_t)(b * 64 + t) * 36 + i] * at[i];
    gs[t] = g;
  }
  __syncthreads();
  if (t < 7) {
    float o = fc3b[t];
    for (int c = 0; c < 64; ++c) o += gs[c] * fc3w[t * 64 + c];
    out[b * 7 + t] = o;
  }
}

// ---------------- launcher ----------------
extern "C" void kernel_launch(void* const* d_in, const int* in_sizes, int n_in,
                              void* d_out, int out_size, void* d_ws, size_t ws_size,
                              hipStream_t stream) {
  (void)in_sizes; (void)n_in; (void)out_size; (void)ws_size;
  const float* x    = (const float*)d_in[0];
  const float* c1w  = (const float*)d_in[1];
  const float* c1b  = (const float*)d_in[2];
  const float* bn1g = (const float*)d_in[3];
  const float* bn1b = (const float*)d_in[4];
  const float* bn1m = (const float*)d_in[5];
  const float* bn1v = (const float*)d_in[6];
  const float* c2w  = (const float*)d_in[7];
  const float* c2b  = (const float*)d_in[8];
  const float* bn2g = (const float*)d_in[9];
  const float* bn2b = (const float*)d_in[10];
  const float* bn2m = (const float*)d_in[11];
  const float* bn2v = (const float*)d_in[12];
  const float* c3w  = (const float*)d_in[13];
  const float* c3b  = (const float*)d_in[14];
  const float* bn3g = (const float*)d_in[15];
  const float* bn3b = (const float*)d_in[16];
  const float* bn3m = (const float*)d_in[17];
  const float* bn3v = (const float*)d_in[18];
  const float* fc1w = (const float*)d_in[19];
  const float* fc1b = (const float*)d_in[20];
  const float* fc2w = (const float*)d_in[21];
  const float* fc2b = (const float*)d_in[22];
  const float* attw = (const float*)d_in[23];
  const float* attb = (const float*)d_in[24];
  const float* fc3w = (const float*)d_in[25];
  const float* fc3b = (const float*)d_in[26];
  float* out = (float*)d_out;

  char* p = (char*)d_ws;
  auto alloc = [&](size_t bytes) -> void* {
    void* r = (void*)p;
    p += (bytes + 255) & ~(size_t)255;
    return r;
  };
  __bf16* act1  = (__bf16*)alloc((size_t)512 * 24 * 24 * 256 * 2);  // NHWC
  __bf16* act2  = (__bf16*)alloc((size_t)512 * 12 * 12 * 128 * 2);  // NHWC
  float*  out3f = (float*) alloc((size_t)512 * 64 * 36 * 4);        // NCHW
  __bf16* out3b = (__bf16*)alloc((size_t)512 * 64 * 36 * 2);        // NCHW flat
  __bf16* w2b   = (__bf16*)alloc((size_t)128 * 2304 * 2);           // [N][K]
  __bf16* w3b   = (__bf16*)alloc((size_t)64 * 1152 * 2);
  __bf16* fw1   = (__bf16*)alloc((size_t)512 * 2304 * 2);
  __bf16* fw2   = (__bf16*)alloc((size_t)256 * 512 * 2);
  __bf16* fwa   = (__bf16*)alloc((size_t)64 * 256 * 2);
  __bf16* fc1o  = (__bf16*)alloc((size_t)512 * 512 * 2);
  __bf16* fc2o  = (__bf16*)alloc((size_t)512 * 256 * 2);
  float*  qf    = (float*) alloc((size_t)512 * 64 * 4);
  float*  s1 = (float*)alloc(256 * 4); float* h1 = (float*)alloc(256 * 4);
  float*  s2 = (float*)alloc(128 * 4); float* h2 = (float*)alloc(128 * 4);
  float*  s3 = (float*)alloc(64 * 4);  float* h3 = (float*)alloc(64 * 4);

  // BN folds
  bn_prep<<<1, 256, 0, stream>>>(bn1g, bn1b, bn1m, bn1v, c1b, s1, h1, 256);
  bn_prep<<<1, 128, 0, stream>>>(bn2g, bn2b, bn2m, bn2v, c2b, s2, h2, 128);
  bn_prep<<<1, 64, 0, stream>>>(bn3g, bn3b, bn3m, bn3v, c3b, s3, h3, 64);
  // Weight transforms to bf16 B-matrix layouts [N][K]
  conv_w_prep<<<(128 * 256 * 9 + 255) / 256, 256, 0, stream>>>(c2w, w2b, 128, 256);
  conv_w_prep<<<(64 * 128 * 9 + 255) / 256, 256, 0, stream>>>(c3w, w3b, 64, 128);
  cast_bf16<<<(512 * 2304 + 255) / 256, 256, 0, stream>>>(fc1w, fw1, 512 * 2304);
  cast_bf16<<<(256 * 512 + 255) / 256, 256, 0, stream>>>(fc2w, fw2, 256 * 512);
  cast_bf16<<<(64 * 256 + 255) / 256, 256, 0, stream>>>(attw, fwa, 64 * 256);

  // Block 1: direct conv (1 input channel) + BN + ReLU + pool -> act1 NHWC bf16
  conv1_block_kernel<<<512 * 24 * 24, 256, 0, stream>>>(x, c1w, s1, h1, act1);

  // Block 2: implicit-GEMM WMMA conv + fused BN/ReLU/pool -> act2 NHWC bf16
  gemm_wmma<1, 1><<<dim3(294912 / BM, 128 / BN), 256, 0, stream>>>(
      act1, w2b, s2, h2, act2, nullptr, 294912, 128, 2304, 24, 24, 256, 12, 12);

  // Block 3: implicit-GEMM WMMA conv + fused BN/ReLU/pool -> out3 NCHW (f32 + bf16)
  gemm_wmma<1, 2><<<dim3(73728 / BM, 64 / BN), 256, 0, stream>>>(
      act2, w3b, s3, h3, out3b, out3f, 73728, 64, 1152, 12, 12, 128, 6, 6);

  // FC1 / FC2 / attention projection (dense WMMA GEMMs)
  gemm_wmma<0, 0><<<dim3(512 / BM, 512 / BN), 256, 0, stream>>>(
      out3b, fw1, nullptr, fc1b, fc1o, nullptr, 512, 512, 2304, 1, 1, 1, 1, 1);
  gemm_wmma<0, 0><<<dim3(512 / BM, 256 / BN), 256, 0, stream>>>(
      fc1o, fw2, nullptr, fc2b, fc2o, nullptr, 512, 256, 512, 1, 1, 1, 1, 1);
  gemm_wmma<0, 0><<<dim3(512 / BM, 64 / BN), 256, 0, stream>>>(
      fc2o, fwa, nullptr, attb, nullptr, qf, 512, 64, 256, 1, 1, 1, 1, 1);

  // Spatial attention softmax + fc3
  attn_kernel<<<512, 64, 0, stream>>>(out3f, qf, fc3w, fc3b, out);
}
